// ConvolutionOneWay_3315714752870
// MI455X (gfx1250) — compile-verified, matrix operand order checked by reading, not run on hardware
//
#include <hip/hip_runtime.h>
#include <math.h>

typedef __attribute__((ext_vector_type(16))) _Float16 v16h;
typedef __attribute__((ext_vector_type(8)))  _Float16 v8h;
typedef __attribute__((ext_vector_type(4)))  _Float16 v4h;
typedef __attribute__((ext_vector_type(8)))  float    v8f;
typedef __attribute__((ext_vector_type(4)))  unsigned int u32x4;
typedef __attribute__((ext_vector_type(8)))  int      i32x8;
typedef __attribute__((ext_vector_type(4)))  int      i32x4;

#define DEV static __device__ __forceinline__

DEV v8f wmma_f16(v16h a, v16h b, v8f c) {
  return __builtin_amdgcn_wmma_f32_16x16x32_f16(false, a, false, b, (short)0, c, false, false);
}

DEV float lane_bcast(float v, int srcLane) {
  return __builtin_bit_cast(float,
      __builtin_amdgcn_ds_bpermute(srcLane << 2, __builtin_bit_cast(int, v)));
}

DEV unsigned lds_byte_off(const void* p) { return (unsigned)(unsigned long long)p; }

// ---- TDM: 1-D contiguous copy global -> LDS (data_size=8B). nbytes % 8 == 0,
// nbytes/8 <= 65535 (16-bit tile_dim0). Issued by the calling wave only.
DEV void tdm_copy_global_to_lds(const void* gsrc, unsigned lds_off, unsigned nbytes) {
#if __has_builtin(__builtin_amdgcn_tensor_load_to_lds)
  unsigned long long ga = (unsigned long long)gsrc;
  unsigned n8 = nbytes >> 3;
  // D# group0: count=1, lds_addr, global_addr[56:0], type=2 ("image")
  u32x4 g0 = { 1u, lds_off, (unsigned)ga, (unsigned)(ga >> 32) | 0x80000000u };
  // D# group1: wg_mask=0, data_size=3(8B), tensor_dim0=n8, tensor_dim1=1,
  //            tile_dim0=n8, tile_dim1/2 unused, tensor_dim0_stride=n8
  i32x8 g1 = { (int)(3u << 16),
               (int)((n8 & 0xFFFFu) << 16),
               (int)((n8 >> 16) | (1u << 16)),
               (int)((n8 & 0xFFFFu) << 16),
               0,
               (int)n8, 0, 0 };
  i32x4 z4 = {0, 0, 0, 0};
#if defined(__clang_major__) && __clang_major__ >= 23
  i32x8 z8 = {0,0,0,0,0,0,0,0};
  __builtin_amdgcn_tensor_load_to_lds(g0, g1, z4, z4, z8, 0);
#else
  __builtin_amdgcn_tensor_load_to_lds(g0, g1, z4, z4, 0);
#endif
#if __has_builtin(__builtin_amdgcn_s_wait_tensorcnt)
  __builtin_amdgcn_s_wait_tensorcnt(0);
#endif
#else
  (void)gsrc; (void)lds_off; (void)nbytes;
#endif
}

// Stage a weight panel into LDS (all threads of the block call; barrier inside).
DEV void stage_to_lds(const void* g, void* ldsdst, unsigned nbytes, int tid, int nthr) {
#if __has_builtin(__builtin_amdgcn_tensor_load_to_lds)
  (void)nthr;
  if (tid < 32) tdm_copy_global_to_lds(g, lds_byte_off(ldsdst), nbytes);
#else
  const uint4* s = (const uint4*)g; uint4* d = (uint4*)ldsdst;
  for (unsigned i = (unsigned)tid; i < nbytes / 16u; i += (unsigned)nthr) d[i] = s[i];
#endif
  __syncthreads();
}

// ============================ weight packing ============================
// Packs f32 [K x 128] matrix into the WMMA f16 B-operand layout:
// flat half index i = (((kk*8 + nn)*32) + lane)*16 + e, with
// K index j(e,hs) = e + 8*hs + (e>=8 ? 8 : 0) (hs = lane>>4), K = kk*32 + j,
// column = nn*16 + (lane&15). Normalization scale folded in.
__global__ void pack_fctp_kernel(const float* __restrict__ W, _Float16* __restrict__ dst,
                                 int total, float scale) {
  int i = blockIdx.x * blockDim.x + threadIdx.x;
  if (i >= total) return;
  int e = i & 15, l = (i >> 4) & 31, nn = (i >> 9) & 7, kk = i >> 12;
  int hs = l >> 4;
  int j = e + 8 * hs + ((e >= 8) ? 8 : 0);
  int k = kk * 32 + j;
  int col = nn * 16 + (l & 15);
  dst[i] = (_Float16)(W[k * 128 + col] * scale);
}

// A-layout companion table for W_lin3 (angle dot product): [kk][hs][16] f32.
__global__ void pack_w3_kernel(const float* __restrict__ W3, float* __restrict__ dst,
                               float scale) {
  int i = blockIdx.x * blockDim.x + threadIdx.x;
  if (i >= 1024) return;
  int e = i & 15, hs = (i >> 4) & 1, kk = i >> 5;
  int j = e + 8 * hs + ((e >= 8) ? 8 : 0);
  dst[i] = W3[kk * 32 + j] * scale;
}

__global__ void zero_kernel(float* __restrict__ p, int n) {
  int i = (blockIdx.x * blockDim.x + threadIdx.x) * 4;
  if (i < n) { float4 z = {0.f, 0.f, 0.f, 0.f}; *(float4*)(p + i) = z; }
}

// ============================ node fctp (sender) ============================
// out[n,:] = (x[n,:] (x) y[n,:]) @ Bpk   (scale folded into Bpk)
#define SND_WAVES 4
__global__ __launch_bounds__(32 * SND_WAVES) void fctp_node_kernel(
    const float* __restrict__ xin, const float* __restrict__ yat,
    const v16h* __restrict__ Bpk, float* __restrict__ outf, int N) {
  __shared__ v16h Bsh[8192];                 // 256 KB full [1024 x 128] f16 panel
  __shared__ float xt[SND_WAVES][2048];      // 16 nodes x 128 ch per wave
  const int tid = threadIdx.x;
  const int l = tid & 31, w = tid >> 5;
  const int hs = l >> 4, lm = l & 15;
  stage_to_lds(Bpk, Bsh, sizeof(Bsh), tid, SND_WAVES * 32);
  const int ntiles = (N + 15) >> 4;
  for (int t = blockIdx.x * SND_WAVES + w; t < ntiles; t += gridDim.x * SND_WAVES) {
    const int nbase = t << 4;
    const float4* x4 = (const float4*)xin;
    const int mx = N * 32 - 1;
#pragma unroll
    for (int i = 0; i < 16; ++i) {
      int idx = nbase * 32 + i * 32 + l; idx = idx > mx ? mx : idx;
      ((float4*)xt[w])[i * 32 + l] = x4[idx];
    }
    int nme = nbase + lm; nme = nme < N ? nme : N - 1;
    float4 y0 = ((const float4*)yat)[nme * 2 + 0];
    float4 y1 = ((const float4*)yat)[nme * 2 + 1];
    float yv[8] = {y0.x, y0.y, y0.z, y0.w, y1.x, y1.y, y1.z, y1.w};
    const float* xrow = &xt[w][lm * 128];
    v8f acc[8] = {};
#pragma unroll 1
    for (int kk = 0; kk < 32; ++kk) {
      float xa = xrow[kk * 4 + hs];
      float xb = xrow[kk * 4 + 2 + hs];
      v16h a;
#pragma unroll
      for (int v = 0; v < 8; ++v) {
        a[v] = (_Float16)(xa * yv[v]);
        a[v + 8] = (_Float16)(xb * yv[v]);
      }
#pragma unroll
      for (int nn = 0; nn < 8; ++nn)
        acc[nn] = wmma_f16(a, Bsh[(kk * 8 + nn) * 32 + l], acc[nn]);
    }
    float* orow = outf + (size_t)(nbase + 8 * hs) * 128 + lm;
    if (nbase + 16 <= N) {
      // full tile: unguarded, uniform-branch fast path
#pragma unroll
      for (int nn = 0; nn < 8; ++nn)
#pragma unroll
        for (int r = 0; r < 8; ++r)
          orow[(size_t)r * 128 + nn * 16] = acc[nn][r];
    } else {
#pragma unroll
      for (int nn = 0; nn < 8; ++nn)
#pragma unroll
        for (int r = 0; r < 8; ++r) {
          int n = nbase + r + 8 * hs;
          if (n < N) outf[(size_t)n * 128 + nn * 16 + lm] = acc[nn][r];
        }
    }
  }
}

// ============================ fused edge kernel ============================
// weight = silu-MLP(edge_scalars); ef = weight * sf[src] * eattr; atomic scatter.
__global__ __launch_bounds__(256) void edge_kernel(
    const float* __restrict__ es, const float* __restrict__ ea,
    const int* __restrict__ esrc, const int* __restrict__ edst,
    const float* __restrict__ sf, const v16h* __restrict__ Wfcpk,
    float* __restrict__ rf, int E) {
  __shared__ v16h Wsh[1536];                 // 48 KB: fc1 (512 v16h) + fc2 (1024 v16h)
  __shared__ _Float16 esH[8][1024];          // per-wave 16x64 f16
  __shared__ _Float16 hh[8][2048];           // per-wave 16x128 f16 hidden
  const int tid = threadIdx.x;
  const int l = tid & 31, w = tid >> 5;
  const int hs = l >> 4, lm = l & 15;
  stage_to_lds(Wfcpk, Wsh, sizeof(Wsh), tid, 256);
  const int ntiles = (E + 15) >> 4;
  for (int t = blockIdx.x * 8 + w; t < ntiles; t += gridDim.x * 8) {
    const int eb = t << 4;
    const float4* es4 = (const float4*)es;
    const int mx = E * 16 - 1;
#pragma unroll
    for (int i = 0; i < 8; ++i) {
      int idx = eb * 16 + i * 32 + l; idx = idx > mx ? mx : idx;
      float4 q = es4[idx];
      v4h h4; h4[0] = (_Float16)q.x; h4[1] = (_Float16)q.y;
      h4[2] = (_Float16)q.z; h4[3] = (_Float16)q.w;
      *(v4h*)&esH[w][(i * 32 + l) * 4] = h4;
    }
    // GEMM1: [16,64] @ fc1 -> [16,128]
    v8f c1[8] = {};
#pragma unroll
    for (int kk = 0; kk < 2; ++kk) {
      v8h lo = *(const v8h*)&esH[w][lm * 64 + kk * 32 + 8 * hs];
      v8h hi = *(const v8h*)&esH[w][lm * 64 + kk * 32 + 16 + 8 * hs];
      v16h a;
#pragma unroll
      for (int e2 = 0; e2 < 8; ++e2) { a[e2] = lo[e2]; a[e2 + 8] = hi[e2]; }
#pragma unroll
      for (int nn = 0; nn < 8; ++nn)
        c1[nn] = wmma_f16(a, Wsh[(kk * 8 + nn) * 32 + l], c1[nn]);
    }
    // silu -> hh (row-major f16); SILU_NORM folded into fc2 pack
#pragma unroll
    for (int nn = 0; nn < 8; ++nn)
#pragma unroll
      for (int r = 0; r < 8; ++r) {
        float x = c1[nn][r];
        float s = x / (1.f + __expf(-x));
        hh[w][(r + 8 * hs) * 128 + nn * 16 + lm] = (_Float16)s;
      }
    // GEMM2: [16,128] @ fc2 -> per-edge weight [16,128]
    v8f c2[8] = {};
#pragma unroll
    for (int kk = 0; kk < 4; ++kk) {
      v8h lo = *(const v8h*)&hh[w][lm * 128 + kk * 32 + 8 * hs];
      v8h hi = *(const v8h*)&hh[w][lm * 128 + kk * 32 + 16 + 8 * hs];
      v16h a;
#pragma unroll
      for (int e2 = 0; e2 < 8; ++e2) { a[e2] = lo[e2]; a[e2 + 8] = hi[e2]; }
#pragma unroll
      for (int nn = 0; nn < 8; ++nn)
        c2[nn] = wmma_f16(a, Wsh[512 + (kk * 8 + nn) * 32 + l], c2[nn]);
    }
    // edge meta (8 rows per lane, C layout: row = r + 8*hs)
    int si[8], di[8]; float ev[8];
#pragma unroll
    for (int r = 0; r < 8; ++r) {
      int m = eb + r + 8 * hs; int mc = m < E ? m : E - 1;
      si[r] = esrc[mc]; di[r] = edst[mc]; ev[r] = ea[mc];
    }
    // gather sender_features, multiply, scatter-add
    if (eb + 16 <= E) {
      // full tile: unguarded fast path (one uniform branch)
#pragma unroll
      for (int nn = 0; nn < 8; ++nn) {
        int c = nn * 16 + lm;
#pragma unroll
        for (int r = 0; r < 8; ++r) {
          float g = sf[(size_t)si[r] * 128 + c];
          float val = c2[nn][r] * g * ev[r];
          (void)__hip_atomic_fetch_add(&rf[(size_t)di[r] * 128 + c], val,
                                       __ATOMIC_RELAXED, __HIP_MEMORY_SCOPE_AGENT);
        }
      }
    } else {
#pragma unroll
      for (int nn = 0; nn < 8; ++nn) {
        int c = nn * 16 + lm;
#pragma unroll
        for (int r = 0; r < 8; ++r) {
          int m = eb + r + 8 * hs;
          if (m < E) {
            float g = sf[(size_t)si[r] * 128 + c];
            float val = c2[nn][r] * g * ev[r];
            (void)__hip_atomic_fetch_add(&rf[(size_t)di[r] * 128 + c], val,
                                         __ATOMIC_RELAXED, __HIP_MEMORY_SCOPE_AGENT);
          }
        }
      }
    }
  }
}

// ============================ final receiver kernel ============================
// conv = (rf (x) rat) @ Blin2 ; sc = (rin (x) rat) @ Bsc ; angle = (rf (x) rat)·w3
// out = cos(angle)*sc + sin(angle)*conv
__global__ __launch_bounds__(64) void fctp_final_kernel(
    const float* __restrict__ rin, const float* __restrict__ rat,
    const float* __restrict__ rfeat,
    const v16h* __restrict__ BpkSc, const v16h* __restrict__ BpkL2,
    const float* __restrict__ w3pk, float* __restrict__ outf, int N) {
  __shared__ v16h Bsh[8192];            // lin2 panel, 256 KB (TDM)
  __shared__ float xf[2][2048];
  __shared__ float xi[2][2048];
  const int tid = threadIdx.x;
  const int l = tid & 31, w = tid >> 5;
  const int hs = l >> 4, lm = l & 15;
  stage_to_lds(BpkL2, Bsh, sizeof(Bsh), tid, 64);
  const int ntiles = (N + 15) >> 4;
  const float4* w34 = (const float4*)w3pk;
  for (int t = blockIdx.x * 2 + w; t < ntiles; t += gridDim.x * 2) {
    const int nbase = t << 4;
    const int mx = N * 32 - 1;
    const float4* a4 = (const float4*)rfeat;
    const float4* b4 = (const float4*)rin;
#pragma unroll
    for (int i = 0; i < 16; ++i) {
      int idx = nbase * 32 + i * 32 + l; idx = idx > mx ? mx : idx;
      ((float4*)xf[w])[i * 32 + l] = a4[idx];
      ((float4*)xi[w])[i * 32 + l] = b4[idx];
    }
    int nme = nbase + lm; nme = nme < N ? nme : N - 1;
    float4 y0 = ((const float4*)rat)[nme * 2 + 0];
    float4 y1 = ((const float4*)rat)[nme * 2 + 1];
    float yv[8] = {y0.x, y0.y, y0.z, y0.w, y1.x, y1.y, y1.z, y1.w};
    const float* xrF = &xf[w][lm * 128];
    const float* xrI = &xi[w][lm * 128];
    v8f accC[8] = {}; v8f accS[8] = {};
    float angp = 0.f;
#pragma unroll 1
    for (int kk = 0; kk < 32; ++kk) {
      float xa = xrF[kk * 4 + hs], xb = xrF[kk * 4 + 2 + hs];
      float4 wA = w34[kk * 8 + hs * 4 + 0];
      float4 wB = w34[kk * 8 + hs * 4 + 1];
      float4 wC = w34[kk * 8 + hs * 4 + 2];
      float4 wD = w34[kk * 8 + hs * 4 + 3];
      float w3a[8] = {wA.x, wA.y, wA.z, wA.w, wB.x, wB.y, wB.z, wB.w};
      float w3b[8] = {wC.x, wC.y, wC.z, wC.w, wD.x, wD.y, wD.z, wD.w};
      v16h a1;
#pragma unroll
      for (int v = 0; v < 8; ++v) {
        float pa = xa * yv[v], pb = xb * yv[v];
        angp += pa * w3a[v] + pb * w3b[v];
        a1[v] = (_Float16)pa; a1[v + 8] = (_Float16)pb;
      }
      float xc = xrI[kk * 4 + hs], xd = xrI[kk * 4 + 2 + hs];
      v16h a2;
#pragma unroll
      for (int v = 0; v < 8; ++v) {
        a2[v] = (_Float16)(xc * yv[v]);
        a2[v + 8] = (_Float16)(xd * yv[v]);
      }
#pragma unroll
      for (int nn = 0; nn < 8; ++nn) {
        accC[nn] = wmma_f16(a1, Bsh[(kk * 8 + nn) * 32 + l], accC[nn]);
        accS[nn] = wmma_f16(a2, BpkSc[(kk * 8 + nn) * 32 + l], accS[nn]);
      }
    }
    float tot = angp + lane_bcast(angp, l ^ 16);   // lanes m and m+16 hold node m halves
    float cs[8], sn[8];
#pragma unroll
    for (int r = 0; r < 8; ++r) {
      float ang = lane_bcast(tot, 8 * hs + r);
      cs[r] = __cosf(ang); sn[r] = __sinf(ang);
    }
    float* orow = outf + (size_t)(nbase + 8 * hs) * 128 + lm;
    if (nbase + 16 <= N) {
      // full tile: unguarded fast path
#pragma unroll
      for (int nn = 0; nn < 8; ++nn)
#pragma unroll
        for (int r = 0; r < 8; ++r)
          orow[(size_t)r * 128 + nn * 16] = cs[r] * accS[nn][r] + sn[r] * accC[nn][r];
    } else {
#pragma unroll
      for (int nn = 0; nn < 8; ++nn)
#pragma unroll
        for (int r = 0; r < 8; ++r) {
          int n = nbase + r + 8 * hs;
          if (n < N)
            outf[(size_t)n * 128 + nn * 16 + lm] = cs[r] * accS[nn][r] + sn[r] * accC[nn][r];
        }
    }
  }
}

// ============================ launcher ============================
extern "C" void kernel_launch(void* const* d_in, const int* in_sizes, int n_in,
                              void* d_out, int out_size, void* d_ws, size_t ws_size,
                              hipStream_t stream) {
  (void)n_in; (void)out_size; (void)ws_size;
  const float* sender_input   = (const float*)d_in[0];
  const float* sender_attr    = (const float*)d_in[1];
  const float* receiver_input = (const float*)d_in[2];
  const float* receiver_attr  = (const float*)d_in[3];
  const int*   edge_src       = (const int*)d_in[4];
  const int*   edge_dst       = (const int*)d_in[5];
  const float* edge_attr      = (const float*)d_in[6];
  const float* edge_scalars   = (const float*)d_in[7];
  const float* W_sc   = (const float*)d_in[8];
  const float* W_lin1 = (const float*)d_in[9];
  const float* W_fc1  = (const float*)d_in[10];
  const float* W_fc2  = (const float*)d_in[11];
  const float* W_lin2 = (const float*)d_in[12];
  const float* W_lin3 = (const float*)d_in[13];

  const int Ns = in_sizes[0] / 128;
  const int Nr = in_sizes[2] / 128;
  const int E  = in_sizes[4];

  float* ws = (float*)d_ws;
  size_t off = 0;
  float* sf = ws + off; off += (size_t)Ns * 128;     // sender_features
  float* rf = ws + off; off += (size_t)Nr * 128;     // scatter accumulator
  off = (off + 63) & ~(size_t)63;
  _Float16* pkLin1 = (_Float16*)(ws + off); off += 65536;   // 131072 halves
  _Float16* pkSc   = (_Float16*)(ws + off); off += 65536;
  _Float16* pkLin2 = (_Float16*)(ws + off); off += 65536;
  _Float16* pkFc   = (_Float16*)(ws + off); off += 12288;   // fc1(8192h)+fc2(16384h), contiguous
  float* w3pk = ws + off; off += 1024;

  const float sF  = 1.f / 32.f;                  // 1/sqrt(128*8)
  const float sL2 = 1.f / (32.f * sqrtf(32.f));  // also folds 1/sqrt(NUM_NEIGHBORS)
  pack_fctp_kernel<<<512, 256, 0, stream>>>(W_lin1, pkLin1, 131072, sF);
  pack_fctp_kernel<<<512, 256, 0, stream>>>(W_sc,   pkSc,   131072, sF);
  pack_fctp_kernel<<<512, 256, 0, stream>>>(W_lin2, pkLin2, 131072, sL2);
  pack_fctp_kernel<<<32,  256, 0, stream>>>(W_fc1, pkFc, 8192, 0.125f);  // 1/sqrt(64)
  pack_fctp_kernel<<<64,  256, 0, stream>>>(W_fc2, pkFc + 8192, 16384,
                                            1.679177f / sqrtf(128.f));   // SILU_NORM/sqrt(H)
  pack_w3_kernel<<<4, 256, 0, stream>>>(W_lin3, w3pk, 0.1f * sL2);

  const int nrd = Nr * 128;
  zero_kernel<<<(nrd / 4 + 255) / 256, 256, 0, stream>>>(rf, nrd);

  fctp_node_kernel<<<1024, 32 * SND_WAVES, 0, stream>>>(
      sender_input, sender_attr, (const v16h*)pkLin1, sf, Ns);

  edge_kernel<<<1024, 256, 0, stream>>>(
      edge_scalars, edge_attr, edge_src, edge_dst, sf, (const v16h*)pkFc, rf, E);

  fctp_final_kernel<<<1024, 64, 0, stream>>>(
      receiver_input, receiver_attr, rf, (const v16h*)pkSc, (const v16h*)pkLin2,
      w3pk, (float*)d_out, Nr);
}
